// BLSTMModel_63771674411089
// MI455X (gfx1250) — compile-verified, matrix operand order checked by reading, not run on hardware
//
#include <hip/hip_runtime.h>
#include <math.h>

// ---------------------------------------------------------------------------
// Bidirectional 2-layer LSTM LM for MI455X (gfx1250), bf16 WMMA implementation
// B=256, T=128, VS=64, EMB=256, HS=512
// ---------------------------------------------------------------------------

typedef __attribute__((ext_vector_type(16))) __bf16 v16bf;
typedef __attribute__((ext_vector_type(8)))  __bf16 v8bf;
typedef __attribute__((ext_vector_type(8)))  float  v8f;

constexpr int B_   = 256;
constexpr int T_   = 128;
constexpr int VS_  = 64;
constexpr int EMB_ = 256;
constexpr int HS_  = 512;

union AFrag { v16bf v; v8bf h[2]; };

__device__ __forceinline__ float sigmoidf_(float x) {
    return 1.0f / (1.0f + __expf(-x));
}

// ---------------------------------------------------------------------------
// Weight swizzle: W [N][K] f32 row-major  ->  bf16 WMMA B-fragments.
// Fragment (ct, kt) holds B[k][n] = W[n][k] for n in [ct*16,..), k in [kt*32,..)
// laid out so lane l reads 16 contiguous bf16 (32 bytes):
//   n = ct*16 + (l&15); elements 0..7 -> K = kt*32 + 8*(l>>4) + e
//                       elements 8..15 -> K = kt*32 + 8*(l>>4) + 16 + (e-8)
// (mirrors the 16-bit A-matrix 16x32 layout, ISA 7.12.2)
// ---------------------------------------------------------------------------
__global__ __launch_bounds__(256) void swz_kernel(const float* __restrict__ W,
                                                  __bf16* __restrict__ dst,
                                                  int N, int K) {
    const int lane = threadIdx.x & 31;
    const int frag = blockIdx.x * 8 + (threadIdx.x >> 5);
    const int ktt  = K >> 5;
    const int nt   = N >> 4;
    if (frag >= nt * ktt) return;
    const int ct = frag / ktt;
    const int kt = frag - ct * ktt;
    const int nn = ct * 16 + (lane & 15);
    const int kb = kt * 32 + 8 * (lane >> 4);
    const float* srow = W + (size_t)nn * K;
    __bf16* d = dst + (size_t)frag * 512 + lane * 16;
#pragma unroll
    for (int e = 0; e < 8; ++e) d[e]     = (__bf16)srow[kb + e];
#pragma unroll
    for (int e = 0; e < 8; ++e) d[8 + e] = (__bf16)srow[kb + 16 + e];
}

// ---------------------------------------------------------------------------
// One 16-wide HS column tile of all four gates: D = A x B + bias, then LSTM
// cell update. A = concat(src0[16 x KSPLIT*32], src1[16 x (KT-KSPLIT)*32]).
// ---------------------------------------------------------------------------
template <int KT, int KSPLIT, int S0, int S1>
__device__ __forceinline__ void lstm_gate_tile(const __bf16* __restrict__ src0,
                                               const __bf16* __restrict__ src1,
                                               const __bf16* __restrict__ wfrag,
                                               const float*  __restrict__ bias,
                                               int ctg, int lane,
                                               float* cstate, float* hstash) {
    const int n  = lane & 15;
    const int hi = lane >> 4;
    const int m  = n;            // A row owned by this lane
    const int kb = hi * 8;       // K sub-offset per A layout
    const int col = ctg * 16 + n;

    v8f ai{}, af{}, ag{}, ao{};
    const float bi = bias[col];
    const float bf = bias[HS_     + col];
    const float bg = bias[2 * HS_ + col];
    const float bo = bias[3 * HS_ + col];
#pragma unroll
    for (int r = 0; r < 8; ++r) { ai[r] = bi; af[r] = bf; ag[r] = bg; ao[r] = bo; }

    const __bf16* wl = wfrag + (size_t)lane * 16;
    for (int kt = 0; kt < KT; ++kt) {
        const __bf16* arow = (kt < KSPLIT) ? (src0 + m * S0 + kt * 32)
                                           : (src1 + m * S1 + (kt - KSPLIT) * 32);
        AFrag a;
        a.h[0] = *reinterpret_cast<const v8bf*>(arow + kb);
        a.h[1] = *reinterpret_cast<const v8bf*>(arow + kb + 16);

        v16bf bw0 = *reinterpret_cast<const v16bf*>(wl + (size_t)((0  + ctg) * KT + kt) * 512);
        v16bf bw1 = *reinterpret_cast<const v16bf*>(wl + (size_t)((32 + ctg) * KT + kt) * 512);
        v16bf bw2 = *reinterpret_cast<const v16bf*>(wl + (size_t)((64 + ctg) * KT + kt) * 512);
        v16bf bw3 = *reinterpret_cast<const v16bf*>(wl + (size_t)((96 + ctg) * KT + kt) * 512);

        ai = __builtin_amdgcn_wmma_f32_16x16x32_bf16(false, a.v, false, bw0, (short)0, ai, false, false);
        af = __builtin_amdgcn_wmma_f32_16x16x32_bf16(false, a.v, false, bw1, (short)0, af, false, false);
        ag = __builtin_amdgcn_wmma_f32_16x16x32_bf16(false, a.v, false, bw2, (short)0, ag, false, false);
        ao = __builtin_amdgcn_wmma_f32_16x16x32_bf16(false, a.v, false, bw3, (short)0, ao, false, false);
    }

#pragma unroll
    for (int r = 0; r < 8; ++r) {
        const float iv = sigmoidf_(ai[r]);
        const float fv = sigmoidf_(af[r]);
        const float gv = tanhf(ag[r]);
        const float ov = sigmoidf_(ao[r]);
        const float c  = fv * cstate[r] + iv * gv;
        cstate[r] = c;
        hstash[r] = ov * tanhf(c);
    }
}

// ---------------------------------------------------------------------------
// Persistent LSTM kernel: one workgroup per (direction, 16-row batch tile).
// 8 waves; wave w owns HS column tiles {4w..4w+3}. c-state in VGPRs, h in LDS.
// ---------------------------------------------------------------------------
__global__ __launch_bounds__(256) void blstm_dir_kernel(
    const int*    __restrict__ x,   const float* __restrict__ emb,
    const __bf16* __restrict__ wf0, const __bf16* __restrict__ wf1,
    const __bf16* __restrict__ wb0, const __bf16* __restrict__ wb1,
    const float*  __restrict__ bf0, const float* __restrict__ bf1,
    const float*  __restrict__ bb0, const float* __restrict__ bb1,
    __bf16* __restrict__ hf, __bf16* __restrict__ hb) {

    __shared__ __bf16 s_x [16 * EMB_];  //  8 KB: embedded input tile (bf16)
    __shared__ __bf16 s_h0[16 * HS_];   // 16 KB: layer-0 hidden state
    __shared__ __bf16 s_h1[16 * HS_];   // 16 KB: layer-1 hidden state

    const int dir  = blockIdx.y;
    const int b0   = blockIdx.x * 16;
    const int tid  = threadIdx.x;
    const int wave = tid >> 5;
    const int lane = tid & 31;
    const int n    = lane & 15;
    const int hi   = lane >> 4;

    const __bf16* w0  = dir ? wb0 : wf0;
    const __bf16* w1  = dir ? wb1 : wf1;
    const float*  b0v = dir ? bb0 : bf0;
    const float*  b1v = dir ? bb1 : bf1;
    __bf16* hout = dir ? hb : hf;

    for (int i = tid; i < 16 * HS_; i += 256) {
        s_h0[i] = (__bf16)0.0f;
        s_h1[i] = (__bf16)0.0f;
    }

    float c0[4][8], c1[4][8], hst[4][8];
#pragma unroll
    for (int ct = 0; ct < 4; ++ct)
#pragma unroll
        for (int r = 0; r < 8; ++r) { c0[ct][r] = 0.0f; c1[ct][r] = 0.0f; }

    for (int tt = 0; tt < T_; ++tt) {
        const int t = dir ? (T_ - 1 - tt) : tt;

        __syncthreads();   // h-state init / prior-step reads complete
        // embedding gather: 16 rows x 256 cols, f32 -> bf16
        {
            const int r  = tid >> 4;
            const int cs = (tid & 15) * 16;
            const int tok = x[(b0 + r) * T_ + t];
            const float4* src = reinterpret_cast<const float4*>(emb + (size_t)tok * EMB_ + cs);
#pragma unroll
            for (int i = 0; i < 4; ++i) {
                float4 v = src[i];
                __bf16* d = &s_x[r * EMB_ + cs + i * 4];
                d[0] = (__bf16)v.x; d[1] = (__bf16)v.y;
                d[2] = (__bf16)v.z; d[3] = (__bf16)v.w;
            }
        }
        __syncthreads();

        // ---- layer 0: A = [emb(256) | h0(512)], K = 768 (24 K-tiles) ----
#pragma unroll
        for (int ct = 0; ct < 4; ++ct)
            lstm_gate_tile<24, 8, EMB_, HS_>(s_x, s_h0, w0, b0v,
                                             wave * 4 + ct, lane, c0[ct], hst[ct]);
        __syncthreads();   // all waves done reading s_h0
#pragma unroll
        for (int ct = 0; ct < 4; ++ct) {
            const int colb = (wave * 4 + ct) * 16 + n;
#pragma unroll
            for (int r = 0; r < 8; ++r)
                s_h0[(r + 8 * hi) * HS_ + colb] = (__bf16)hst[ct][r];
        }
        __syncthreads();

        // ---- layer 1: A = [h0(512) | h1(512)], K = 1024 (32 K-tiles) ----
#pragma unroll
        for (int ct = 0; ct < 4; ++ct)
            lstm_gate_tile<32, 16, HS_, HS_>(s_h0, s_h1, w1, b1v,
                                             wave * 4 + ct, lane, c1[ct], hst[ct]);
        __syncthreads();   // all waves done reading s_h1
#pragma unroll
        for (int ct = 0; ct < 4; ++ct) {
            const int colb = (wave * 4 + ct) * 16 + n;
#pragma unroll
            for (int r = 0; r < 8; ++r)
                s_h1[(r + 8 * hi) * HS_ + colb] = (__bf16)hst[ct][r];
        }
        __syncthreads();

        // write layer-1 h tile to global at the UN-flipped time index
        {
            const uint4* src = reinterpret_cast<const uint4*>(s_h1);
            uint4* dst = reinterpret_cast<uint4*>(hout + ((size_t)t * B_ + b0) * HS_);
#pragma unroll
            for (int i = 0; i < 4; ++i) dst[tid + i * 256] = src[tid + i * 256];
        }
    }
}

// ---------------------------------------------------------------------------
// Head: compressed = tanh([fwd|bwd] @ Wc^T + bc); logits = compressed @ Wfc^T + bfc
// One workgroup per (t, 16-row batch tile).
// ---------------------------------------------------------------------------
__global__ __launch_bounds__(256) void head_kernel(
    const __bf16* __restrict__ hf,  const __bf16* __restrict__ hb,
    const __bf16* __restrict__ wc,  const float*  __restrict__ bc,
    const __bf16* __restrict__ wfc, const float*  __restrict__ bfc,
    float* __restrict__ out) {

    __shared__ __bf16 s_comb[16 * 1024];  // 32 KB
    __shared__ __bf16 s_c   [16 * HS_];   // 16 KB

    const int bt = blockIdx.x & 15;
    const int t  = blockIdx.x >> 4;
    const int b0 = bt * 16;
    const int tid  = threadIdx.x;
    const int wave = tid >> 5;
    const int lane = tid & 31;
    const int n  = lane & 15;
    const int hi = lane >> 4;
    const int m  = n;
    const int kb = hi * 8;

    // load fwd|bwd tile into LDS (16 x 1024 bf16)
    {
        const uint4* f = reinterpret_cast<const uint4*>(hf + ((size_t)t * B_ + b0) * HS_);
        const uint4* b = reinterpret_cast<const uint4*>(hb + ((size_t)t * B_ + b0) * HS_);
        uint4* dc = reinterpret_cast<uint4*>(s_comb);
#pragma unroll
        for (int i = 0; i < 4; ++i) {
            const int idx = tid + i * 256;        // 16 rows x 64 uint4 chunks
            const int r = idx >> 6, c = idx & 63;
            dc[r * 128 + c]      = f[idx];
            dc[r * 128 + 64 + c] = b[idx];
        }
    }
    __syncthreads();

    // stage 1: 1024 -> 512, tanh
#pragma unroll
    for (int ct = 0; ct < 4; ++ct) {
        const int ctg = wave * 4 + ct;
        const int col = ctg * 16 + n;
        v8f acc{};
        const float bcv = bc[col];
#pragma unroll
        for (int r = 0; r < 8; ++r) acc[r] = bcv;
        for (int kt = 0; kt < 32; ++kt) {
            AFrag a;
            const __bf16* arow = s_comb + m * 1024 + kt * 32;
            a.h[0] = *reinterpret_cast<const v8bf*>(arow + kb);
            a.h[1] = *reinterpret_cast<const v8bf*>(arow + kb + 16);
            v16bf bw = *reinterpret_cast<const v16bf*>(
                wc + (size_t)(ctg * 32 + kt) * 512 + lane * 16);
            acc = __builtin_amdgcn_wmma_f32_16x16x32_bf16(false, a.v, false, bw,
                                                          (short)0, acc, false, false);
        }
#pragma unroll
        for (int r = 0; r < 8; ++r)
            s_c[(r + 8 * hi) * HS_ + col] = (__bf16)tanhf(acc[r]);
    }
    __syncthreads();

    // stage 2: 512 -> 64 logits (waves 0..3), f32 out [B][T][VS]
    if (wave < 4) {
        const int ct  = wave;
        const int col = ct * 16 + n;
        v8f acc{};
        const float bv = bfc[col];
#pragma unroll
        for (int r = 0; r < 8; ++r) acc[r] = bv;
        for (int kt = 0; kt < 16; ++kt) {
            AFrag a;
            const __bf16* arow = s_c + m * HS_ + kt * 32;
            a.h[0] = *reinterpret_cast<const v8bf*>(arow + kb);
            a.h[1] = *reinterpret_cast<const v8bf*>(arow + kb + 16);
            v16bf bw = *reinterpret_cast<const v16bf*>(
                wfc + (size_t)(ct * 16 + kt) * 512 + lane * 16);
            acc = __builtin_amdgcn_wmma_f32_16x16x32_bf16(false, a.v, false, bw,
                                                          (short)0, acc, false, false);
        }
#pragma unroll
        for (int r = 0; r < 8; ++r)
            out[((size_t)(b0 + r + 8 * hi) * T_ + t) * VS_ + col] = acc[r];
    }
}

// ---------------------------------------------------------------------------
// Host-side launch
// ---------------------------------------------------------------------------
extern "C" void kernel_launch(void* const* d_in, const int* in_sizes, int n_in,
                              void* d_out, int out_size, void* d_ws, size_t ws_size,
                              hipStream_t stream) {
    (void)in_sizes; (void)n_in; (void)out_size; (void)ws_size;

    const int*   x   = (const int*)  d_in[0];
    const float* emb = (const float*)d_in[1];
    const float* Wf0 = (const float*)d_in[2];
    const float* bf0 = (const float*)d_in[3];
    const float* Wf1 = (const float*)d_in[4];
    const float* bf1 = (const float*)d_in[5];
    const float* Wb0 = (const float*)d_in[6];
    const float* bb0 = (const float*)d_in[7];
    const float* Wb1 = (const float*)d_in[8];
    const float* bb1 = (const float*)d_in[9];
    const float* Wc  = (const float*)d_in[10];
    const float* bc  = (const float*)d_in[11];
    const float* Wfc = (const float*)d_in[12];
    const float* bfc = (const float*)d_in[13];
    float* out = (float*)d_out;

    // ws layout (bf16 elements); total ~83 MB
    __bf16* ws = (__bf16*)d_ws;
    size_t o = 0;
    __bf16* wf0 = ws + o; o += (size_t)128 * 24 * 512;  // 2048x768
    __bf16* wf1 = ws + o; o += (size_t)128 * 32 * 512;  // 2048x1024
    __bf16* wb0 = ws + o; o += (size_t)128 * 24 * 512;
    __bf16* wb1 = ws + o; o += (size_t)128 * 32 * 512;
    __bf16* wcc = ws + o; o += (size_t)32  * 32 * 512;  // 512x1024
    __bf16* wfc = ws + o; o += (size_t)4   * 16 * 512;  // 64x512
    __bf16* hfb = ws + o; o += (size_t)T_ * B_ * HS_;   // fwd layer-1 h
    __bf16* hbb = ws + o; o += (size_t)T_ * B_ * HS_;   // bwd layer-1 h

    // 1) swizzle weights into WMMA B-fragment layout (bf16)
    swz_kernel<<<(128 * 24) / 8, 256, 0, stream>>>(Wf0, wf0, 2048, 768);
    swz_kernel<<<(128 * 32) / 8, 256, 0, stream>>>(Wf1, wf1, 2048, 1024);
    swz_kernel<<<(128 * 24) / 8, 256, 0, stream>>>(Wb0, wb0, 2048, 768);
    swz_kernel<<<(128 * 32) / 8, 256, 0, stream>>>(Wb1, wb1, 2048, 1024);
    swz_kernel<<<(32 * 32)  / 8, 256, 0, stream>>>(Wc,  wcc, 512,  1024);
    swz_kernel<<<(4 * 16)   / 8, 256, 0, stream>>>(Wfc, wfc, 64,   512);

    // 2) both directions of the 2-layer LSTM (persistent, recurrent)
    blstm_dir_kernel<<<dim3(16, 2), 256, 0, stream>>>(
        x, emb, wf0, wf1, wb0, wb1, bf0, bf1, bb0, bb1, hfb, hbb);

    // 3) compress + vocab projection
    head_kernel<<<T_ * 16, 256, 0, stream>>>(hfb, hbb, wcc, bc, wfc, bfc, out);
}
